// mymodel_15934328668219
// MI455X (gfx1250) — matrix operand order repaired and checked by reading, not compiled, not run on hardware
//
#include <hip/hip_runtime.h>
#include <hip/hip_bf16.h>
#include <math.h>

// ---------------- constants ----------------
enum : int {
  Cc    = 64,
  HEADS = 4,
  HWs   = 65536,         // H*W
  NB    = 2,
  NPIX  = NB * HWs,      // 131072
  HIDc  = 170,
  HID2  = 340,
  HIDP  = 352,           // M-padded for w_in (mult of 16)
  KGP   = 192            // K-padded gate buffer for w_out (mult of 32)
};

typedef __attribute__((ext_vector_type(16))) __bf16    v16bf;
typedef __attribute__((ext_vector_type(8)))  float     v8f;
typedef __attribute__((ext_vector_type(4)))  unsigned  v4u;
typedef __attribute__((ext_vector_type(8)))  unsigned  v8u;

union V16 { v16bf v; unsigned short s[16]; };

// native f32 -> bf16 (RNE) via hardware convert
static __device__ __forceinline__ unsigned short f2bf(float f) {
  union { __bf16 h; unsigned short u; } cv;
  cv.h = (__bf16)f;
  return cv.u;
}

// ---------------- weight fp32 -> bf16 (zero-padded [Mp][Kp]) ----------------
__global__ void wconv_kernel(const float* __restrict__ src, unsigned short* __restrict__ dst,
                             int M, int Ksrc, int Mp, int Kp, int srcMoff) {
  int i = blockIdx.x * 256 + threadIdx.x;
  if (i >= Mp * Kp) return;
  int m = i / Kp, k = i % Kp;
  float v = (m < M && k < Ksrc) ? src[(size_t)(srcMoff + m) * Ksrc + k] : 0.f;
  dst[i] = f2bf(v);
}

// ---------------- LayerNorm over 64 channels per pixel -> bf16 ----------------
__global__ void ln_kernel(const float* __restrict__ x, const float* __restrict__ g,
                          const float* __restrict__ bta, unsigned short* __restrict__ out) {
  int p = blockIdx.x * 256 + threadIdx.x;
  if (p >= NPIX) return;
  int b = p >> 16, s = p & 65535;
  size_t base = (size_t)b * Cc * HWs + s;
  float sum = 0.f, sq = 0.f;
  for (int c = 0; c < Cc; ++c) {
    float v = x[base + (size_t)c * HWs];
    sum += v; sq += v * v;
  }
  float mu   = sum * (1.f / Cc);
  float var  = sq * (1.f / Cc) - mu * mu;
  float rinv = rsqrtf(var + 1e-5f);
  for (int c = 0; c < Cc; ++c) {
    float v = (x[base + (size_t)c * HWs] - mu) * rinv;
    out[base + (size_t)c * HWs] = f2bf(v * g[c] + bta[c]);
  }
}

// ---------------- WMMA GEMM with TDM (tensor_load_to_lds) B-tile staging ----------------
// Block: one 16-row M-tile x 128 pixels (8 waves x 16 px). B tile [Kp][128] bf16 DMA'd to LDS.
// W: bf16 [Mp][Kp] row-major.  in: bf16 activations, layout (b, Kp, HW).
template <bool HASF, bool HASB, bool HASR>
__global__ void gemm_wmma(const unsigned short* __restrict__ Wb,
                          const unsigned short* __restrict__ inb,
                          float* __restrict__ outf,
                          unsigned short* __restrict__ outb,
                          const float* __restrict__ res,
                          int Mp, int Kp, int Mvalid, int CbufOut, int CbufRes) {
  extern __shared__ unsigned short ldsT[];           // [Kp][128] bf16
  const int NG = NPIX / 128;                         // 1024 pixel groups
  int mt = blockIdx.x / NG, ng = blockIdx.x % NG;
  int nbase = ng * 128;
  int b = nbase >> 16, sb = nbase & 65535;
  int wave = threadIdx.x >> 5, lane = threadIdx.x & 31;
  int l16 = lane & 15, hs = lane >> 4;
  int m0 = mt * 16;

  // One wave issues the Tensor Data Mover descriptor: 2-D tile, dim0 = 128 px (2B elems),
  // dim1 = Kp rows with stride HW, packed contiguously into LDS rows of 128 elements.
  if (threadIdx.x == 0) {
    unsigned long long ga = (unsigned long long)inb +
                            2ull * ((size_t)b * Kp * (size_t)HWs + (size_t)sb);
    unsigned ldso = (unsigned)(unsigned long long)(void*)ldsT;
    v4u g0; v8u g1;
    g0[0] = 1u;                                        // count=1, user descriptor
    g0[1] = ldso;                                      // lds_addr
    g0[2] = (unsigned)ga;                              // global_addr[31:0]
    g0[3] = (unsigned)(ga >> 32) | (2u << 30);         // global_addr[56:32] | type=2
    g1[0] = 1u << 16;                                  // data_size = 2 bytes
    g1[1] = ((unsigned)HWs & 0xFFFFu) << 16;           // tensor_dim0[15:0]
    g1[2] = ((unsigned)HWs >> 16) | ((unsigned)Kp << 16); // dim0[31:16] | dim1[15:0]
    g1[3] = 128u << 16;                                // dim1[31:16]=0 | tile_dim0=128
    g1[4] = (unsigned)Kp;                              // tile_dim1=Kp, tile_dim2=0
    g1[5] = (unsigned)HWs;                             // tensor_dim0_stride[31:0]
    g1[6] = 0u;                                        // stride0[47:32]=0, stride1 lo=0
    g1[7] = 0u;
    asm volatile("tensor_load_to_lds %0, %1" :: "s"(g0), "s"(g1) : "memory");
    __builtin_amdgcn_s_wait_tensorcnt(0);
  }
  __syncthreads();

  const unsigned short* wrow = Wb + (size_t)(m0 + l16) * Kp;
  const unsigned short* bcol = ldsT + (wave * 16 + l16);
  v8f acc = {};
  for (int kk = 0; kk < Kp; kk += 32) {
    V16 A, Bf;
    int ko = kk + hs * 8;                    // A: lanes 0-15: K {kk..+7, kk+16..+23}
#pragma unroll
    for (int j = 0; j < 8; ++j) { A.s[j] = wrow[ko + j]; A.s[8 + j] = wrow[ko + 16 + j]; }
    int kb = kk + hs * 8;                    // B: same K order as A
#pragma unroll
    for (int j = 0; j < 8; ++j) { Bf.s[j] = bcol[(ko + j) * 128]; Bf.s[8 + j] = bcol[(ko + 16 + j) * 128]; }
    acc = __builtin_amdgcn_wmma_f32_16x16x32_bf16(false, A.v, false, Bf.v, (short)0, acc, false, false);
  }
  size_t col = (size_t)sb + wave * 16 + l16;
#pragma unroll
  for (int r = 0; r < 8; ++r) {
    int m = m0 + r + hs * 8;                 // C: vgpr r = rows {r, r+8}
    if (m >= Mvalid) continue;
    float v = acc[r];
    if (HASR) v += res[((size_t)b * CbufRes + m) * HWs + col];
    if (HASF) outf[((size_t)b * CbufOut + m) * HWs + col] = v;
    if (HASB) outb[((size_t)b * CbufOut + m) * HWs + col] = f2bf(v);
  }
}

// ---------------- depthwise 3x3, SAME ----------------
__global__ void dwconv_kernel(const float* __restrict__ in, const float* __restrict__ wt,
                              float* __restrict__ out, int Cbuf) {
  int c = blockIdx.y;
  int p = blockIdx.x * 256 + threadIdx.x;
  int b = p >> 16, s = p & 65535;
  int h = s >> 8, w = s & 255;
  const float* W = wt + c * 9;
  size_t base = ((size_t)b * Cbuf + c) * HWs;
  float acc = 0.f;
#pragma unroll
  for (int dy = -1; dy <= 1; ++dy) {
    int hh = h + dy; if (hh < 0 || hh > 255) continue;
#pragma unroll
    for (int dx = -1; dx <= 1; ++dx) {
      int ww = w + dx; if (ww < 0 || ww > 255) continue;
      acc += in[base + hh * 256 + ww] * W[(dy + 1) * 3 + (dx + 1)];
    }
  }
  out[base + s] = acc;
}

// ---------------- row L2 norms for q (ch 0..63) and k (ch 64..127) of qkv_dw ----------------
__global__ void rownorm_kernel(const float* __restrict__ qkv, float* __restrict__ norms) {
  int b = blockIdx.x >> 7, c = blockIdx.x & 127;
  size_t base = ((size_t)b * 192 + c) * HWs;
  float ss = 0.f;
  for (int s = threadIdx.x; s < HWs; s += 256) { float v = qkv[base + s]; ss += v * v; }
  __shared__ float red[256];
  red[threadIdx.x] = ss; __syncthreads();
  for (int st = 128; st > 0; st >>= 1) {
    if (threadIdx.x < st) red[threadIdx.x] += red[threadIdx.x + st];
    __syncthreads();
  }
  if (threadIdx.x == 0) norms[blockIdx.x] = sqrtf(red[0]);
}

// ---------------- attn_raw[bh][d][e] += q_d . k_e  (WMMA over pixel dim) ----------------
__global__ void attn_qk_kernel(const float* __restrict__ qkv, float* __restrict__ attn_raw) {
  int wave = threadIdx.x >> 5, lane = threadIdx.x & 31;
  int l16 = lane & 15, hs = lane >> 4;
  int gw = blockIdx.x * 8 + wave;            // 512 waves
  int bh = gw >> 6, chunk = gw & 63;         // 64 chunks x 1024 px each
  int b = bh >> 2, hd = bh & 3;
  const float* qrow = qkv + ((size_t)b * 192 + hd * 16 + l16) * HWs;
  const float* krow = qkv + ((size_t)b * 192 + 64 + hd * 16 + l16) * HWs;
  v8f acc = {};
  int n0e = chunk * 1024 + 1024;
  for (int n0 = chunk * 1024; n0 < n0e; n0 += 32) {
    __builtin_prefetch(qrow + n0 + 32, 0, 1);          // unconditional: stays in-buffer
    __builtin_prefetch(krow + n0 + 32, 0, 1);
    V16 A, Bf;
    int ka = n0 + hs * 8;
#pragma unroll
    for (int j = 0; j < 8; ++j) { A.s[j] = f2bf(qrow[ka + j]); A.s[8 + j] = f2bf(qrow[ka + 16 + j]); }
    int kb = n0 + hs * 16;
#pragma unroll
    for (int j = 0; j < 8; ++j) { Bf.s[j] = f2bf(krow[ka + j]); Bf.s[8 + j] = f2bf(krow[ka + 16 + j]); }
    acc = __builtin_amdgcn_wmma_f32_16x16x32_bf16(false, A.v, false, Bf.v, (short)0, acc, false, false);
  }
#pragma unroll
  for (int r = 0; r < 8; ++r)
    atomicAdd(&attn_raw[bh * 256 + (r + hs * 8) * 16 + l16], acc[r]);
}

// ---------------- softmax(QK/(|q||k|) * temp) -> bf16 ----------------
__global__ void softmax_kernel(const float* __restrict__ raw, const float* __restrict__ norms,
                               const float* __restrict__ temp, unsigned short* __restrict__ attn_bf) {
  int bh = blockIdx.x, b = bh >> 2, hd = bh & 3;
  int d = threadIdx.x;
  if (d >= 16) return;
  float nq = fmaxf(norms[b * 128 + hd * 16 + d], 1e-12f);
  float tp = temp[hd];
  float row[16]; float mx = -3.0e38f;
  for (int e = 0; e < 16; ++e) {
    float nk = fmaxf(norms[b * 128 + 64 + hd * 16 + e], 1e-12f);
    float v = raw[bh * 256 + d * 16 + e] / (nq * nk) * tp;
    row[e] = v; mx = fmaxf(mx, v);
  }
  float sum = 0.f;
  for (int e = 0; e < 16; ++e) { row[e] = expf(row[e] - mx); sum += row[e]; }
  float inv = 1.f / sum;
  for (int e = 0; e < 16; ++e) attn_bf[bh * 256 + d * 16 + e] = f2bf(row[e] * inv);
}

// ---------------- out[ch][n] = attn(16x16) @ v  (WMMA, K padded 16->32 w/ zeros) ----------------
__global__ void attn_apply_kernel(const unsigned short* __restrict__ attn_bf,
                                  const float* __restrict__ vsrc,
                                  unsigned short* __restrict__ outb,
                                  int vch0, int CbufV) {
  int wave = threadIdx.x >> 5, lane = threadIdx.x & 31;
  int l16 = lane & 15, hs = lane >> 4;
  int t = blockIdx.x * 8 + wave;             // 32768 waves: 8 bh x 4096 n-tiles
  int nt = t & 4095, bh = t >> 12;
  int b = bh >> 2, hd = bh & 3;
  int s0 = nt * 16;
  V16 A, Bf;
  const unsigned short* arow = attn_bf + bh * 256 + l16 * 16;
#pragma unroll
  for (int j = 0; j < 8; ++j) { A.s[j] = hs ? 0 : arow[j]; A.s[8 + j] = hs ? 0 : arow[8 + j]; }  // K 16..31 (hs==1) zero
  if (hs == 0) {
    size_t base = ((size_t)b * CbufV + vch0) * HWs + (size_t)(s0 + l16);
#pragma unroll
    for (int j = 0; j < 16; ++j) Bf.s[j] = f2bf(vsrc[base + (size_t)j * HWs]);
  } else {
#pragma unroll
    for (int j = 0; j < 16; ++j) Bf.s[j] = 0;                                 // K rows 16..31
  }
  v8f acc = {};
  acc = __builtin_amdgcn_wmma_f32_16x16x32_bf16(false, A.v, false, Bf.v, (short)0, acc, false, false);
#pragma unroll
  for (int r = 0; r < 8; ++r) {
    int m = r + hs * 8;
    outb[((size_t)b * 64 + hd * 16 + m) * HWs + (size_t)(s0 + l16)] = f2bf(acc[r]);
  }
}

// ---------------- GDFN: fused depthwise3x3 (both halves) + exact GELU gate -> bf16 ----------------
__global__ void gdfn_dwgate_kernel(const float* __restrict__ x12, const float* __restrict__ wt,
                                   unsigned short* __restrict__ gbuf) {
  int c = blockIdx.y;                        // 0..191
  int p = blockIdx.x * 256 + threadIdx.x;
  int b = p >> 16, s = p & 65535;
  if (c >= HIDc) { gbuf[((size_t)b * KGP + c) * HWs + s] = 0; return; }
  int h = s >> 8, w = s & 255;
  size_t base1 = ((size_t)b * HIDP + c) * HWs;
  size_t base2 = ((size_t)b * HIDP + HIDc + c) * HWs;
  const float* W1 = wt + c * 9;
  const float* W2 = wt + (HIDc + c) * 9;
  float a = 0.f, g2 = 0.f;
#pragma unroll
  for (int dy = -1; dy <= 1; ++dy) {
    int hh = h + dy; if (hh < 0 || hh > 255) continue;
#pragma unroll
    for (int dx = -1; dx <= 1; ++dx) {
      int ww = w + dx; if (ww < 0 || ww > 255) continue;
      int off = hh * 256 + ww, wi = (dy + 1) * 3 + (dx + 1);
      a  += x12[base1 + off] * W1[wi];
      g2 += x12[base2 + off] * W2[wi];
    }
  }
  float gl = 0.5f * a * (1.f + erff(a * 0.70710678118654752f));   // exact GELU
  gbuf[((size_t)b * KGP + c) * HWs + s] = f2bf(gl * g2);
}

// ---------------- host orchestration ----------------
extern "C" void kernel_launch(void* const* d_in, const int* in_sizes, int n_in,
                              void* d_out, int out_size, void* d_ws, size_t ws_size,
                              hipStream_t stream) {
  (void)in_sizes; (void)n_in; (void)out_size; (void)ws_size;
  const float* x     = (const float*)d_in[0];
  const float* n1w   = (const float*)d_in[1];
  const float* n1b   = (const float*)d_in[2];
  const float* temp  = (const float*)d_in[3];
  const float* wqkv  = (const float*)d_in[4];
  const float* wqkdw = (const float*)d_in[5];
  const float* wpo   = (const float*)d_in[6];
  const float* wkv   = (const float*)d_in[7];
  const float* wkvdw = (const float*)d_in[8];
  const float* wpof  = (const float*)d_in[9];
  const float* n2w   = (const float*)d_in[10];
  const float* n2b   = (const float*)d_in[11];
  const float* win   = (const float*)d_in[12];
  const float* wdw   = (const float*)d_in[13];
  const float* wout  = (const float*)d_in[14];
  float* out = (float*)d_out;

  char* ws = (char*)d_ws;
  size_t o = 0;
  auto alloc = [&](size_t bytes) { void* p = ws + o; o += (bytes + 255) & ~(size_t)255; return p; };

  float*          BIG1   = (float*)alloc((size_t)HIDP * NPIX * 4);   // qkv -> x12
  float*          BIG2   = (float*)alloc((size_t)192  * NPIX * 4);   // qkv_dw -> vf -> gbuf
  unsigned short* LNBF   = (unsigned short*)alloc((size_t)Cc * NPIX * 2);
  unsigned short* POBF   = (unsigned short*)alloc((size_t)Cc * NPIX * 2);
  unsigned short* OUTABF = (unsigned short*)alloc((size_t)Cc * NPIX * 2);
  float*          Y1     = (float*)alloc((size_t)Cc * NPIX * 4);
  float*          NORMS  = (float*)alloc(256 * 4);
  float*          ARAW   = (float*)alloc(8 * 256 * 4);
  unsigned short* ABF    = (unsigned short*)alloc(8 * 256 * 2);
  unsigned short* WQb    = (unsigned short*)alloc((size_t)192 * 64 * 2);
  unsigned short* WPOb   = (unsigned short*)alloc((size_t)64 * 64 * 2);
  unsigned short* WKVb   = (unsigned short*)alloc((size_t)64 * 64 * 2);   // v-slice only
  unsigned short* WPOFb  = (unsigned short*)alloc((size_t)64 * 64 * 2);
  unsigned short* WINb   = (unsigned short*)alloc((size_t)HIDP * 64 * 2);
  unsigned short* WOUTb  = (unsigned short*)alloc((size_t)64 * KGP * 2);

  auto wc = [&](const float* src, unsigned short* dst, int M, int Ksrc, int Mp, int Kp, int moff) {
    int n = Mp * Kp;
    wconv_kernel<<<dim3((n + 255) / 256), dim3(256), 0, stream>>>(src, dst, M, Ksrc, Mp, Kp, moff);
  };
  // variants: F = fp32 out, B = bf16 out, R = +residual
  auto gemmF = [&](const unsigned short* Wb, const unsigned short* inb, float* of,
                   int Mp, int Kp, int Mv, int CO) {
    gemm_wmma<true, false, false><<<dim3((Mp / 16) * (NPIX / 128)), dim3(256),
                                    (size_t)Kp * 128 * 2, stream>>>(
        Wb, inb, of, nullptr, nullptr, Mp, Kp, Mv, CO, 0);
  };
  auto gemmB = [&](const unsigned short* Wb, const unsigned short* inb, unsigned short* ob,
                   int Mp, int Kp, int Mv, int CO) {
    gemm_wmma<false, true, false><<<dim3((Mp / 16) * (NPIX / 128)), dim3(256),
                                    (size_t)Kp * 128 * 2, stream>>>(
        Wb, inb, nullptr, ob, nullptr, Mp, Kp, Mv, CO, 0);
  };
  auto gemmFR = [&](const unsigned short* Wb, const unsigned short* inb, float* of,
                    const float* res, int Mp, int Kp, int Mv, int CO, int CR) {
    gemm_wmma<true, false, true><<<dim3((Mp / 16) * (NPIX / 128)), dim3(256),
                                   (size_t)Kp * 128 * 2, stream>>>(
        Wb, inb, of, nullptr, res, Mp, Kp, Mv, CO, CR);
  };

  // weights -> bf16 (padded)
  wc(wqkv, WQb,   192,  64, 192,  64, 0);
  wc(wpo,  WPOb,   64,  64,  64,  64, 0);
  wc(wkv,  WKVb,   64,  64,  64,  64, 64);   // rows 64..127 (v half; k half is dead code)
  wc(wpof, WPOFb,  64,  64,  64,  64, 0);
  wc(win,  WINb,  HID2, 64, HIDP, 64, 0);
  wc(wout, WOUTb,  64, HIDc, 64, KGP, 0);

  // ---- MDTA ----
  ln_kernel<<<dim3(NPIX / 256), dim3(256), 0, stream>>>(x, n1w, n1b, LNBF);
  gemmF(WQb, LNBF, BIG1, 192, 64, 192, 192);                                  // qkv
  dwconv_kernel<<<dim3(NPIX / 256, 192), dim3(256), 0, stream>>>(BIG1, wqkdw, BIG2, 192);
  rownorm_kernel<<<dim3(256), dim3(256), 0, stream>>>(BIG2, NORMS);
  hipMemsetAsync(ARAW, 0, 8 * 256 * 4, stream);
  attn_qk_kernel<<<dim3(64), dim3(256), 0, stream>>>(BIG2, ARAW);
  softmax_kernel<<<dim3(8), dim3(32), 0, stream>>>(ARAW, NORMS, temp, ABF);
  attn_apply_kernel<<<dim3(4096), dim3(256), 0, stream>>>(ABF, BIG2, OUTABF, 128, 192); // attn @ v
  gemmB(WPOb, OUTABF, POBF, 64, 64, 64, 64);                                  // w_po
  float* VFPRE = BIG2;
  float* VF    = BIG2 + (size_t)Cc * NPIX;
  gemmF(WKVb, POBF, VFPRE, 64, 64, 64, 64);                                   // v half of w_kv
  dwconv_kernel<<<dim3(NPIX / 256, 64), dim3(256), 0, stream>>>(VFPRE, wkvdw + 64 * 9, VF, 64);
  attn_apply_kernel<<<dim3(4096), dim3(256), 0, stream>>>(ABF, VF, OUTABF, 0, 64);      // attn @ vf
  gemmFR(WPOFb, OUTABF, Y1, x, 64, 64, 64, 64, 64);                           // w_pof + residual x

  // ---- GDFN ----
  ln_kernel<<<dim3(NPIX / 256), dim3(256), 0, stream>>>(Y1, n2w, n2b, LNBF);
  gemmF(WINb, LNBF, BIG1, HIDP, 64, HID2, HIDP);                              // w_in (340 rows valid)
  unsigned short* GBUF = (unsigned short*)BIG2;
  gdfn_dwgate_kernel<<<dim3(NPIX / 256, KGP), dim3(256), 0, stream>>>(BIG1, wdw, GBUF);
  gemmFR(WOUTb, GBUF, out, Y1, 64, KGP, 64, 64, 64);                          // w_out + residual y1
}
